// DynGrowingHNN_48550310314285
// MI455X (gfx1250) — compile-verified, matrix operand order checked by reading, not run on hardware
//
#include <hip/hip_runtime.h>
#include <hip/hip_bf16.h>
#include <math.h>

// ---------------------------------------------------------------------------
// DynGrowingHNN on MI455X (gfx1250):
//   - LDS-staged WMMA f16->f32 GEMMs
//   - weight panels streamed with GLOBAL_LOAD_ASYNC_TO_LDS_B128 (ASYNCcnt),
//     double-buffered against WMMA compute
//   - wave-per-edge scatter/gather with f32 global atomics
// wave32 everywhere.
// ---------------------------------------------------------------------------

typedef _Float16 v16h __attribute__((ext_vector_type(16)));
typedef _Float16 v8h  __attribute__((ext_vector_type(8)));
typedef float    v8f  __attribute__((ext_vector_type(8)));

#define HDIM 128
#define MSEG 50000   // hyperedge segment count (reference constant)

__device__ __forceinline__ float sigmoidf_(float x) {
  return 1.0f / (1.0f + __expf(-x));
}

// ---------------------------------------------------------------------------
// Weight prep: W (K x ncols, f32, row-major) -> Wt (ncols x K, f16) so GEMM
// B-panels are contiguous-per-column and can be byte-copied by the async DMA.
// ---------------------------------------------------------------------------
__global__ void wtrans_kernel(const float* __restrict__ W, _Float16* __restrict__ Wt,
                              int K, int ncols)
{
  int idx = blockIdx.x * blockDim.x + threadIdx.x;
  if (idx >= K * ncols) return;
  int k = idx / ncols, c = idx % ncols;
  Wt[(size_t)c * K + k] = (_Float16)W[idx];
}

// ---------------------------------------------------------------------------
// LDS-staged WMMA GEMM:
//   D[rows x NCOLS] = act( [A1 (*1/max(rc1,1)) | A2 (*1/max(rc2,1))] @ W + bias )
//   A1, A2 : rows x 128 f32 (A2 only when NKB==8 -> K=256 concat)
//   Wt     : NCOLS x K f16  (pre-transposed weights); K = NKB*32
// One workgroup = one 16-row block covering ALL column tiles
// (blockDim = (32, NCOLS/16), one 16x16 tile per wave).
//   - A block staged once: 16 x K f16 in LDS (count-scale fused)
//   - W k-panels (NCOLS x 32 f16) streamed global->LDS via async DMA,
//     double-buffered; s_wait_asynccnt + split barrier gate consumption
// The barrier after the last k-panel makes A1==Dst (in-place) safe.
// ---------------------------------------------------------------------------
template <int NKB, int NCOLS>
__global__ void wmma_gemm_kernel(const float* __restrict__ A1, const float* __restrict__ rc1,
                                 const float* __restrict__ A2, const float* __restrict__ rc2,
                                 const _Float16* __restrict__ Wt, const float* __restrict__ bias,
                                 float* __restrict__ Dst, int relu_flag)
{
  constexpr int KTOT     = NKB * 32;
  constexpr int NWAVES   = NCOLS / 16;
  constexpr int NTHREADS = NWAVES * 32;

  __shared__ _Float16 atile[16 * KTOT];        // [row][k] row-major, scaled f16
  __shared__ _Float16 btile[2][NCOLS * 32];    // [col][k] k-panel, double buffer

  const int lane = threadIdx.x;                // 0..31
  const int l    = lane & 15;
  const int hi   = lane >> 4;                  // lane half: K +8/+24 (A,B) / M +8 (C,D)
  const int cb   = threadIdx.y * 16;           // this wave's column-tile base
  const int rb   = blockIdx.x * 16;            // row-tile base
  const int tid  = threadIdx.y * 32 + lane;

  // ---- async-DMA issue of one NCOLS x 32 weight k-panel into btile[buf] ----
  // NCOLS*32 halfs = NCOLS*4 16-byte chunks; NTHREADS threads -> 2 chunks each.
  auto issueB = [&](int kb, int buf) {
#pragma unroll
    for (int q = 0; q < 2; ++q) {
      const int chunk = tid + q * NTHREADS;    // 0 .. NCOLS*4-1
      const int c  = chunk >> 2;               // column
      const int qq = chunk & 3;                // 8-half sub-chunk within panel row
      const _Float16* gp = Wt + (size_t)c * KTOT + kb * 32 + qq * 8;
      const unsigned lds = (unsigned)(uintptr_t)(&btile[buf][c * 32 + qq * 8]);
      asm volatile("global_load_async_to_lds_b128 %0, %1, off"
                   :: "v"(lds), "v"(gp) : "memory");
    }
  };

  // ---- stage A block: 16 x KTOT, f32 -> f16, per-row 1/max(count,1) fused ----
  for (int idx = tid * 4; idx < 16 * KTOT; idx += NTHREADS * 4) {
    const int r = idx / KTOT;                  // compile-time div
    const int k = idx % KTOT;
    const int grow = rb + r;
    const float* Ap; int kk; const float* rc;
    if (NKB == 4 || k < 128) { Ap = A1; kk = k;       rc = rc1; }
    else                     { Ap = A2; kk = k - 128; rc = rc2; }
    const float sc = rc ? (1.0f / fmaxf(rc[grow], 1.0f)) : 1.0f;
    const float4 a = *(const float4*)&Ap[(size_t)grow * HDIM + kk];
    atile[idx + 0] = (_Float16)(a.x * sc);
    atile[idx + 1] = (_Float16)(a.y * sc);
    atile[idx + 2] = (_Float16)(a.z * sc);
    atile[idx + 3] = (_Float16)(a.w * sc);
  }

  issueB(0, 0);                                // prologue: panel 0 in flight

  v8f acc = {};

#pragma unroll
  for (int kb = 0; kb < NKB; ++kb) {
    // (a) previous iteration's fragment reads done before overwriting its buffer
    __syncthreads();

    if (kb + 1 < NKB) {
      issueB(kb + 1, (kb + 1) & 1);            // prefetch next panel
      asm volatile("s_wait_asynccnt 0x2" ::: "memory");  // panel kb complete
    } else {
      asm volatile("s_wait_asynccnt 0x0" ::: "memory");
    }
    // (b) panel kb visible to all waves (and A tile on kb==0)
    __syncthreads();

    // ---- fragments from LDS: two contiguous 8-half runs per lane ----
    const v8h* ap = (const v8h*)(atile + l * KTOT + kb * 32 + hi * 8);
    const v8h a0 = ap[0];
    const v8h a1 = ap[2];                      // +16 halfs
    const v8h* bp = (const v8h*)(&btile[kb & 1][(cb + l) * 32 + hi * 8]);
    const v8h b0 = bp[0];
    const v8h b1 = bp[2];

    const v16h af = __builtin_shufflevector(a0, a1, 0, 1, 2, 3, 4, 5, 6, 7,
                                            8, 9, 10, 11, 12, 13, 14, 15);
    const v16h bf = __builtin_shufflevector(b0, b1, 0, 1, 2, 3, 4, 5, 6, 7,
                                            8, 9, 10, 11, 12, 13, 14, 15);

    acc = __builtin_amdgcn_wmma_f32_16x16x32_f16(false, af, false, bf,
                                                 (short)0, acc, false, false);
  }

  __syncthreads();   // all A/W reads complete before (possibly in-place) stores

  const float bc = bias ? bias[cb + l] : 0.0f;
#pragma unroll
  for (int r = 0; r < 8; ++r) {
    float v = acc[r] + bc;
    if (relu_flag) v = fmaxf(v, 0.0f);
    // C/D layout: VGPR r -> M=r (lanes 0-15) / M=r+8 (lanes 16-31); lane -> N
    Dst[(size_t)(rb + r + hi * 8) * NCOLS + cb + l] = v;
  }
}

// ---------------------------------------------------------------------------
// Per-edge degree counts for both edge types in one pass.
// ---------------------------------------------------------------------------
__global__ void edge_count_kernel(const int* __restrict__ ridx, const int* __restrict__ cidx,
                                  const int* __restrict__ attr,
                                  float* __restrict__ ce0, float* __restrict__ cn0,
                                  float* __restrict__ ce1, float* __restrict__ cn1, int E)
{
  int e = blockIdx.x * blockDim.x + threadIdx.x;
  if (e >= E) return;
  if (attr[e] == 0) {
    atomicAdd(&ce0[cidx[e]], 1.0f);
    atomicAdd(&cn0[ridx[e]], 1.0f);
  } else {
    atomicAdd(&ce1[cidx[e]], 1.0f);
    atomicAdd(&cn1[ridx[e]], 1.0f);
  }
}

// ---------------------------------------------------------------------------
// One wave per incidence entry: gather a 128-float row from src[gidx[e]],
// scatter-add into dst[sidx[e]] (f32 global atomics). Edge-type filter is
// wave-uniform so EXEC stays full for the vector path.
// ---------------------------------------------------------------------------
__global__ void edge_scatter_kernel(const float* __restrict__ src, const int* __restrict__ gidx,
                                    const int* __restrict__ sidx, const int* __restrict__ attr,
                                    int type, float* __restrict__ dst, int E)
{
  int wid = blockIdx.x * (blockDim.x >> 5) + (threadIdx.x >> 5);
  if (wid >= E) return;
  if (attr[wid] != type) return;   // wave-uniform
  int lane = threadIdx.x & 31;

  const float* srow = src + (size_t)gidx[wid] * HDIM;
  float*       drow = dst + (size_t)sidx[wid] * HDIM;
  __builtin_prefetch(drow + lane * 4, 1, 1);           // global_prefetch_b8

  float4 v = ((const float4*)srow)[lane];
  float* d = drow + lane * 4;
  atomicAdd(d + 0, v.x);
  atomicAdd(d + 1, v.y);
  atomicAdd(d + 2, v.z);
  atomicAdd(d + 3, v.w);
}

// ---------------------------------------------------------------------------
// GRU elementwise (PyTorch semantics). gi/gh are rows x 384 (bias already in).
// ---------------------------------------------------------------------------
__global__ void gru_kernel(const float* __restrict__ gi, const float* __restrict__ gh,
                           const float* __restrict__ hprev, float* __restrict__ hnext, int N)
{
  int idx = blockIdx.x * blockDim.x + threadIdx.x;
  if (idx >= N * HDIM) return;
  int row = idx >> 7, c = idx & 127;
  const float* gir = gi + (size_t)row * 384;
  const float* ghr = gh + (size_t)row * 384;
  float r = sigmoidf_(gir[c]       + ghr[c]);
  float z = sigmoidf_(gir[128 + c] + ghr[128 + c]);
  float n = tanhf(gir[256 + c] + r * ghr[256 + c]);
  hnext[idx] = (1.0f - z) * n + z * hprev[idx];
}

// ---------------------------------------------------------------------------
// out = (h_next @ Wro + bro)[:, :3]  -- only 3 columns needed, plain VALU.
// ---------------------------------------------------------------------------
__global__ void readout_kernel(const float* __restrict__ hn, const float* __restrict__ Wro,
                               const float* __restrict__ bro, float* __restrict__ out, int N)
{
  int row = blockIdx.x * blockDim.x + threadIdx.x;
  if (row >= N) return;
  float o0 = 0.f, o1 = 0.f, o2 = 0.f;
  const float* h = hn + (size_t)row * HDIM;
#pragma unroll 4
  for (int k = 0; k < HDIM; ++k) {
    float a = h[k];
    const float* w = Wro + (size_t)k * HDIM;
    o0 += a * w[0]; o1 += a * w[1]; o2 += a * w[2];
  }
  float* o = out + (size_t)row * 3;
  o[0] = o0 + bro[0]; o[1] = o1 + bro[1]; o[2] = o2 + bro[2];
}

// ---------------------------------------------------------------------------
extern "C" void kernel_launch(void* const* d_in, const int* in_sizes, int n_in,
                              void* d_out, int out_size, void* d_ws, size_t ws_size,
                              hipStream_t stream)
{
  (void)n_in; (void)out_size; (void)ws_size;

  const float* x     = (const float*)d_in[0];
  const float* hprev = (const float*)d_in[1];
  const float* Wv0   = (const float*)d_in[2];
  const float* We0   = (const float*)d_in[3];
  const float* Wu0   = (const float*)d_in[4];
  const float* bu0   = (const float*)d_in[5];
  const float* Wv1   = (const float*)d_in[6];
  const float* We1   = (const float*)d_in[7];
  const float* Wu1   = (const float*)d_in[8];
  const float* bu1   = (const float*)d_in[9];
  const float* Wmix  = (const float*)d_in[10];
  const float* bmix  = (const float*)d_in[11];
  const float* Wih   = (const float*)d_in[12];
  const float* Whh   = (const float*)d_in[13];
  const float* bih   = (const float*)d_in[14];
  const float* bhh   = (const float*)d_in[15];
  const float* Wro   = (const float*)d_in[16];
  const float* bro   = (const float*)d_in[17];
  const int*   eidx  = (const int*)d_in[18];
  const int*   eattr = (const int*)d_in[19];

  const int N = in_sizes[1] / HDIM;   // 50000
  const int E = in_sizes[19];         // 1000000
  const int* row = eidx;              // edge_index[0]
  const int* col = eidx + E;          // edge_index[1]

  const size_t NH = (size_t)N * HDIM;
  const size_t MH = (size_t)MSEG * HDIM;

  float* ws  = (float*)d_ws;
  float* xp  = ws;  ws += NH;                 // node projection x@Wv (per type, reused)
  float* es  = ws;  ws += MH;                 // e_sum, then e_proj in-place
  float* na  = ws;  ws += NH;                 // n_agg accumulator
  float* o0  = ws;  ws += NH;                 // conv output type 0
  float* o1  = ws;  ws += NH;                 // conv output type 1
  float* hb  = ws;  ws += NH;                 // mixed hidden h
  float* gi  = ws;  ws += (size_t)N * 384;    // GRU input gates
  float* gh  = ws;  ws += (size_t)N * 384;    // GRU hidden gates
  float* ce0 = ws;  ws += MSEG;               // counts (contiguous for one memset)
  float* cn0 = ws;  ws += N;
  float* ce1 = ws;  ws += MSEG;
  float* cn1 = ws;  ws += N;

  // pre-transposed f16 weights ([col][K]) for async-DMA B panels
  _Float16* wt    = (_Float16*)ws;
  _Float16* Wv0t  = wt;  wt += 128 * 128;
  _Float16* We0t  = wt;  wt += 128 * 128;
  _Float16* Wu0t  = wt;  wt += 256 * 128;
  _Float16* Wv1t  = wt;  wt += 128 * 128;
  _Float16* We1t  = wt;  wt += 128 * 128;
  _Float16* Wu1t  = wt;  wt += 256 * 128;
  _Float16* Wmixt = wt;  wt += 256 * 128;
  _Float16* Wiht  = wt;  wt += 128 * 384;
  _Float16* Whht  = wt;  wt += 128 * 384;

  // ---- weight prep (tiny, L2-resident) ----
  wtrans_kernel<<<(128 * 128 + 255) / 256, 256, 0, stream>>>(Wv0,  Wv0t,  128, 128);
  wtrans_kernel<<<(128 * 128 + 255) / 256, 256, 0, stream>>>(We0,  We0t,  128, 128);
  wtrans_kernel<<<(256 * 128 + 255) / 256, 256, 0, stream>>>(Wu0,  Wu0t,  256, 128);
  wtrans_kernel<<<(128 * 128 + 255) / 256, 256, 0, stream>>>(Wv1,  Wv1t,  128, 128);
  wtrans_kernel<<<(128 * 128 + 255) / 256, 256, 0, stream>>>(We1,  We1t,  128, 128);
  wtrans_kernel<<<(256 * 128 + 255) / 256, 256, 0, stream>>>(Wu1,  Wu1t,  256, 128);
  wtrans_kernel<<<(256 * 128 + 255) / 256, 256, 0, stream>>>(Wmix, Wmixt, 256, 128);
  wtrans_kernel<<<(128 * 384 + 255) / 256, 256, 0, stream>>>(Wih,  Wiht,  128, 384);
  wtrans_kernel<<<(128 * 384 + 255) / 256, 256, 0, stream>>>(Whh,  Whht,  128, 384);

  // ---- degree counts ----
  hipMemsetAsync(ce0, 0, (2 * (size_t)MSEG + 2 * (size_t)N) * sizeof(float), stream);
  edge_count_kernel<<<(E + 255) / 256, 256, 0, stream>>>(row, col, eattr,
                                                         ce0, cn0, ce1, cn1, E);

  const int rowTiles = N / 16;       // 3125 (exact)
  const int mTiles   = MSEG / 16;    // 3125 (exact)
  const dim3 b128(32, HDIM / 16);    // 8 waves: 16 rows x 128 cols per WG
  const dim3 b384(32, 384 / 16);     // 24 waves: 16 rows x 384 cols per WG
  const int scatterGrid = (E + 7) / 8;  // 8 waves per 256-thread block

  for (int t = 0; t < 2; ++t) {
    const _Float16* Wvt = t == 0 ? Wv0t : Wv1t;
    const _Float16* Wet = t == 0 ? We0t : We1t;
    const _Float16* Wut = t == 0 ? Wu0t : Wu1t;
    const float*    bu  = t == 0 ? bu0  : bu1;
    float* ce = t == 0 ? ce0 : ce1;
    float* cn = t == 0 ? cn0 : cn1;
    float* ot = t == 0 ? o0  : o1;

    // xp = x @ Wv
    wmma_gemm_kernel<4, 128><<<rowTiles, b128, 0, stream>>>(
        x, nullptr, nullptr, nullptr, Wvt, nullptr, xp, 0);
    // e_sum = segment_sum(xp[row] masked by type, col)
    hipMemsetAsync(es, 0, MH * sizeof(float), stream);
    edge_scatter_kernel<<<scatterGrid, 256, 0, stream>>>(xp, row, col, eattr, t, es, E);
    // e_proj = (e_sum / max(cnt_e,1)) @ We   (in place over es)
    wmma_gemm_kernel<4, 128><<<mTiles, b128, 0, stream>>>(
        es, ce, nullptr, nullptr, Wet, nullptr, es, 0);
    // n_sum = segment_sum(e_proj[col] masked by type, row)
    hipMemsetAsync(na, 0, NH * sizeof(float), stream);
    edge_scatter_kernel<<<scatterGrid, 256, 0, stream>>>(es, col, row, eattr, t, na, E);
    // out_t = relu([xp | n_sum/max(cnt_n,1)] @ Wu + bu)
    wmma_gemm_kernel<8, 128><<<rowTiles, b128, 0, stream>>>(
        xp, nullptr, na, cn, Wut, bu, ot, 1);
  }

  // h = relu([out0 | out1] @ Wmix + bmix)
  wmma_gemm_kernel<8, 128><<<rowTiles, b128, 0, stream>>>(
      o0, nullptr, o1, nullptr, Wmixt, bmix, hb, 1);

  // GRU gate pre-activations
  wmma_gemm_kernel<4, 384><<<rowTiles, b384, 0, stream>>>(
      hb, nullptr, nullptr, nullptr, Wiht, bih, gi, 0);
  wmma_gemm_kernel<4, 384><<<rowTiles, b384, 0, stream>>>(
      hprev, nullptr, nullptr, nullptr, Whht, bhh, gh, 0);

  // h_next -> d_out[0 : N*H], out -> d_out[N*H : N*H + N*3]
  float* hnext = (float*)d_out;
  float* outro = hnext + NH;
  gru_kernel<<<((int)NH + 255) / 256, 256, 0, stream>>>(gi, gh, hprev, hnext, N);
  readout_kernel<<<(N + 255) / 256, 256, 0, stream>>>(hnext, Wro, bro, outro, N);
}
